// LSTM_63402307223694
// MI455X (gfx1250) — compile-verified, matrix operand order checked by reading, not run on hardware
//
#include <hip/hip_runtime.h>
#include <hip/hip_bf16.h>

typedef __attribute__((ext_vector_type(16))) __bf16 v16bf;
typedef __attribute__((ext_vector_type(8)))  __bf16 v8bf;
typedef __attribute__((ext_vector_type(8)))  float  v8f;

#define HN 1024
#define TN 256
#define BN 64
#define DIN0 256

__device__ __forceinline__ __bf16 f2bf(float f) {
    union { float f; unsigned u; } v; v.f = f;
    unsigned r = v.u + 0x7FFFu + ((v.u >> 16) & 1u);   // RNE
    unsigned short h = (unsigned short)(r >> 16);
    return __builtin_bit_cast(__bf16, h);
}

__device__ __forceinline__ float sigf(float x) {
    return 1.0f / (1.0f + __expf(-x));
}

// WGP-scope prefetch: pulls into ALL cache levels (default scope).
// SE/SYS-scope prefetches (what __builtin_prefetch produces) stop at GL2,
// which is useless here since the weights are already L2-resident.
__device__ __forceinline__ void prefetch_wgp(const void* p) {
    asm volatile("global_prefetch_b8 %0, off" :: "v"(p));
}

// ---------------------------------------------------------------------------
// Repack W[N4=4096, K] (f32 row-major) -> bf16 WMMA B-fragment order:
// packed[(nt*kTiles + kt)*512 + lane*16 + j]
//   lane: n_local = lane&15, hi = lane>>4
//   j<8 : k = kt*32 + 8*hi + j           (VGPRs 0..3, packed pairs)
//   j>=8: k = kt*32 + 16 + 8*hi + (j-8)  (VGPRs 4..7)
// ---------------------------------------------------------------------------
__global__ __launch_bounds__(256) void repack_w(const float* __restrict__ W,
                                                __bf16* __restrict__ out, int K) {
    const int kTiles = K >> 5;
    const int tid  = blockIdx.x * 256 + threadIdx.x;
    const int lane = tid & 31;
    const int tile = tid >> 5;
    const int kt = tile % kTiles;
    const int nt = tile / kTiles;
    if (nt >= 256) return;                       // 4096/16 n-tiles
    const int nglob = nt * 16 + (lane & 15);
    const int hi    = lane >> 4;
    const float* row = W + (size_t)nglob * K + kt * 32 + 8 * hi;
    v8bf c0, c1;
#pragma unroll
    for (int j = 0; j < 8; ++j) {
        c0[j] = f2bf(row[j]);
        c1[j] = f2bf(row[16 + j]);
    }
    v8bf* dst = (v8bf*)(out + ((size_t)tile * 32 + lane) * 16);
    dst[0] = c0;
    dst[1] = c1;
}

__global__ __launch_bounds__(256) void bias_comb(const float* __restrict__ bi,
                                                 const float* __restrict__ bh,
                                                 float* __restrict__ out, int n) {
    int i = blockIdx.x * 256 + threadIdx.x;
    if (i < n) out[i] = bi[i] + bh[i];
}

__global__ __launch_bounds__(256) void cvt_f32_bf16(const float* __restrict__ in,
                                                    __bf16* __restrict__ out, int n) {
    int i = blockIdx.x * 256 + threadIdx.x;
    if (i < n) out[i] = f2bf(in[i]);
}

__global__ __launch_bounds__(256) void init_hc(__bf16* __restrict__ h,
                                               float* __restrict__ c, int n) {
    int i = blockIdx.x * 256 + threadIdx.x;
    if (i < n) {
        h[i] = __builtin_bit_cast(__bf16, (unsigned short)0);
        c[i] = 0.0f;
    }
}

// ---------------------------------------------------------------------------
// One LSTM timestep, fused: gates = hIn @ Whh^T + x_t @ Wih^T + bias, then
// elementwise update. Block = 16 batch rows x 32 hidden cols; 8 waves = 4
// gates x 2 n-tiles, each wave one 16x16 WMMA tile.
// ---------------------------------------------------------------------------
__global__ __launch_bounds__(256) void lstm_step_kernel(
    const __bf16* __restrict__ hIn,   // [B,H] bf16
    __bf16* __restrict__ hOut,        // [B,H] bf16 (ping-pong, no aliasing)
    const __bf16* __restrict__ xSeq,  // [B,T,Din] bf16
    __bf16* __restrict__ ySeq,        // [B,T,H] bf16 (this layer's output)
    const __bf16* __restrict__ pWih,  // packed, kTiles = Din/32
    const __bf16* __restrict__ pWhh,  // packed, kTiles = H/32
    const float* __restrict__ bias,   // [4H] f32 (b_ih + b_hh)
    float* __restrict__ cBuf,         // [B,H] f32
    float* __restrict__ hF32,         // [B,H] f32 (for FC)
    int t, int Din)
{
    __shared__ float sg[4][16][32];

    const int tid  = threadIdx.x;
    const int lane = tid & 31;
    const int wave = tid >> 5;
    const int gate = wave >> 1;
    const int half = wave & 1;
    const int mt = blockIdx.x & 3;        // batch tile (B=64 -> 4)
    const int ng = blockIdx.x >> 2;       // 32-col hidden group (H/32 = 32)

    const int hi   = lane >> 4;
    const int bRow = mt * 16 + (lane & 15);
    const int ntileG = gate * (HN / 16) + ng * 2 + half;  // tile index in 4H

    v8f acc = {};

    // ---- recurrent contraction: K over H ----
    {
        const __bf16* aRow = hIn + (size_t)bRow * HN;
        const __bf16* bBase = pWhh + (size_t)ntileG * (HN / 32) * 512 + (size_t)lane * 16;
#pragma unroll 4
        for (int kt = 0; kt < HN / 32; ++kt) {
            const __bf16* ap = aRow + kt * 32 + 8 * hi;
            v8bf a0 = *(const v8bf*)(ap);
            v8bf a1 = *(const v8bf*)(ap + 16);
            const __bf16* bp = bBase + (size_t)kt * 512;
            v8bf b0 = *(const v8bf*)(bp);
            v8bf b1 = *(const v8bf*)(bp + 8);
            prefetch_wgp(bp + 512);              // next k-tile into WGP$/L0
            v16bf av = __builtin_shufflevector(a0, a1, 0,1,2,3,4,5,6,7,8,9,10,11,12,13,14,15);
            v16bf bv = __builtin_shufflevector(b0, b1, 0,1,2,3,4,5,6,7,8,9,10,11,12,13,14,15);
            acc = __builtin_amdgcn_wmma_f32_16x16x32_bf16(
                false, av, false, bv, (short)0, acc, false, false);
        }
    }

    // ---- input contraction: K over Din ----
    {
        const __bf16* aRow = xSeq + ((size_t)bRow * TN + t) * Din;
        const int kT2 = Din >> 5;
        const __bf16* bBase = pWih + (size_t)ntileG * kT2 * 512 + (size_t)lane * 16;
#pragma unroll 4
        for (int kt = 0; kt < kT2; ++kt) {
            const __bf16* ap = aRow + kt * 32 + 8 * hi;
            v8bf a0 = *(const v8bf*)(ap);
            v8bf a1 = *(const v8bf*)(ap + 16);
            const __bf16* bp = bBase + (size_t)kt * 512;
            v8bf b0 = *(const v8bf*)(bp);
            v8bf b1 = *(const v8bf*)(bp + 8);
            prefetch_wgp(bp + 512);              // next k-tile into WGP$/L0
            v16bf av = __builtin_shufflevector(a0, a1, 0,1,2,3,4,5,6,7,8,9,10,11,12,13,14,15);
            v16bf bv = __builtin_shufflevector(b0, b1, 0,1,2,3,4,5,6,7,8,9,10,11,12,13,14,15);
            acc = __builtin_amdgcn_wmma_f32_16x16x32_bf16(
                false, av, false, bv, (short)0, acc, false, false);
        }
    }

    // ---- scatter C fragments to LDS: lanes<16 -> M=r, lanes>=16 -> M=8+r ----
#pragma unroll
    for (int r = 0; r < 8; ++r)
        sg[gate][r + 8 * hi][half * 16 + (lane & 15)] = acc[r];
    __syncthreads();

    // ---- elementwise LSTM update: 16x32 = 512 elements, 2 per thread ----
#pragma unroll
    for (int e = tid; e < 512; e += 256) {
        const int m  = e >> 5;
        const int nl = e & 31;
        const int b  = mt * 16 + m;
        const int n  = ng * 32 + nl;
        float xi = sg[0][m][nl] + bias[n];
        float xf = sg[1][m][nl] + bias[HN + n];
        float xg = sg[2][m][nl] + bias[2 * HN + n];
        float xo = sg[3][m][nl] + bias[3 * HN + n];
        const size_t off = (size_t)b * HN + n;
        float cp = cBuf[off];
        float cn = sigf(xf) * cp + sigf(xi) * tanhf(xg);
        float hn = sigf(xo) * tanhf(cn);
        cBuf[off] = cn;
        __bf16 hb = f2bf(hn);
        hOut[off] = hb;
        ySeq[((size_t)b * TN + t) * HN + n] = hb;
        hF32[off] = hn;
    }
}

// ---------------------------------------------------------------------------
// Final FC: out[b] = dot(hF32[b,:], fc_w) + fc_b   (D_OUT = 1)
// ---------------------------------------------------------------------------
__global__ __launch_bounds__(256) void fc_kernel(const float* __restrict__ hF,
                                                 const float* __restrict__ w,
                                                 const float* __restrict__ bptr,
                                                 float* __restrict__ out) {
    __shared__ float red[256];
    const int b = blockIdx.x;
    float s = 0.0f;
    for (int n = threadIdx.x; n < HN; n += 256)
        s += hF[(size_t)b * HN + n] * w[n];
    red[threadIdx.x] = s;
    __syncthreads();
    for (int off = 128; off > 0; off >>= 1) {
        if (threadIdx.x < off) red[threadIdx.x] += red[threadIdx.x + off];
        __syncthreads();
    }
    if (threadIdx.x == 0) out[b] = red[0] + bptr[0];
}

extern "C" void kernel_launch(void* const* d_in, const int* in_sizes, int n_in,
                              void* d_out, int out_size, void* d_ws, size_t ws_size,
                              hipStream_t stream) {
    (void)in_sizes; (void)n_in; (void)out_size; (void)ws_size;
    const float* x = (const float*)d_in[0];
    const float* w_ih[3] = {(const float*)d_in[1], (const float*)d_in[5], (const float*)d_in[9]};
    const float* w_hh[3] = {(const float*)d_in[2], (const float*)d_in[6], (const float*)d_in[10]};
    const float* b_ih[3] = {(const float*)d_in[3], (const float*)d_in[7], (const float*)d_in[11]};
    const float* b_hh[3] = {(const float*)d_in[4], (const float*)d_in[8], (const float*)d_in[12]};
    const float* fc_w = (const float*)d_in[13];
    const float* fc_b = (const float*)d_in[14];
    float* out = (float*)d_out;

    // ---- carve workspace ----
    char* p = (char*)d_ws;
    auto carve = [&](size_t bytes) -> char* {
        char* r = p; p += (bytes + 255) & ~(size_t)255; return r;
    };
    __bf16* pWhh[3];
    __bf16* pWih[3];
    float*  biasC[3];
    for (int l = 0; l < 3; ++l) pWhh[l] = (__bf16*)carve((size_t)4 * HN * HN * 2);
    pWih[0] = (__bf16*)carve((size_t)4 * HN * DIN0 * 2);
    pWih[1] = (__bf16*)carve((size_t)4 * HN * HN * 2);
    pWih[2] = (__bf16*)carve((size_t)4 * HN * HN * 2);
    for (int l = 0; l < 3; ++l) biasC[l] = (float*)carve((size_t)4 * HN * 4);
    __bf16* y0 = (__bf16*)carve((size_t)BN * TN * DIN0 * 2);   // layer-0 input (x as bf16)
    __bf16* yA = (__bf16*)carve((size_t)BN * TN * HN * 2);
    __bf16* yB = (__bf16*)carve((size_t)BN * TN * HN * 2);
    __bf16* hPing = (__bf16*)carve((size_t)BN * HN * 2);
    __bf16* hPong = (__bf16*)carve((size_t)BN * HN * 2);
    float* cBuf = (float*)carve((size_t)BN * HN * 4);
    float* hF32 = (float*)carve((size_t)BN * HN * 4);

    // ---- weight repack + bias combine + input convert ----
    for (int l = 0; l < 3; ++l) {
        const int K1 = (l == 0) ? DIN0 : HN;
        const int blkIh = (256 * (K1 / 32) * 32) / 256;
        repack_w<<<blkIh, 256, 0, stream>>>(w_ih[l], pWih[l], K1);
        repack_w<<<(256 * (HN / 32) * 32) / 256, 256, 0, stream>>>(w_hh[l], pWhh[l], HN);
        bias_comb<<<(4 * HN + 255) / 256, 256, 0, stream>>>(b_ih[l], b_hh[l], biasC[l], 4 * HN);
    }
    {
        const int n = BN * TN * DIN0;
        cvt_f32_bf16<<<(n + 255) / 256, 256, 0, stream>>>(x, y0, n);
    }

    // ---- three LSTM layers, 256 timesteps each ----
    const __bf16* xin[3]  = {y0, yA, yB};
    __bf16*       yout[3] = {yA, yB, yA};
    const int     din[3]  = {DIN0, HN, HN};
    for (int l = 0; l < 3; ++l) {
        init_hc<<<(BN * HN + 255) / 256, 256, 0, stream>>>(hPing, cBuf, BN * HN);
        __bf16* hI = hPing;
        __bf16* hO = hPong;
        for (int t = 0; t < TN; ++t) {
            lstm_step_kernel<<<128, 256, 0, stream>>>(
                hI, hO, xin[l], yout[l], pWih[l], pWhh[l], biasC[l],
                cBuf, hF32, t, din[l]);
            __bf16* tmp = hI; hI = hO; hO = tmp;
        }
    }

    // ---- final FC ----
    fc_kernel<<<BN, 256, 0, stream>>>(hF32, fc_w, fc_b, out);
}
